// BiLSTM_CRF_6270652252792
// MI455X (gfx1250) — matrix operand order repaired and checked by reading, not compiled, not run on hardware
//
#include <hip/hip_runtime.h>
#include <hip/hip_bf16.h>
#include <math.h>

// ---------------------------------------------------------------------------
// BiLSTM-CRF for MI455X (gfx1250, wave32, WMMA).
// Pipeline: (1) pack weights to bf16 WMMA B-fragment layout + bias sums,
//           (2) init logits to b_out,
//           (3) persistent recurrent kernel: 4 WGs (dir x batch-half), WMMA
//               bf16 GEMMs per step, fused LSTM cell + output projection,
//           (4) Viterbi decode, one wave per sequence, backpointers in LDS.
// Round 3: branch-free activations (tanh(x) = 2*sigmoid(2x)-1) to shorten the
//          serial per-step tail; projection dot unrolled. GEMM loop unchanged
//          (round-2 asm showed a clean pipelined WMMA loop, no spills).
// ---------------------------------------------------------------------------

typedef __attribute__((ext_vector_type(16))) __bf16 v16bf;
typedef __attribute__((ext_vector_type(8)))  float  v8f;

union FragAB { unsigned int u[8]; v16bf v; };

static constexpr int PER  = 128 * 1024;   // dwords per packed weight matrix
static constexpr int NLOG = 32 * 1024 * 14;

__device__ __forceinline__ unsigned short f2bf(float f) {
  unsigned int u = __float_as_uint(f);
  u += 0x7FFFu + ((u >> 16) & 1u);         // round-to-nearest-even
  return (unsigned short)(u >> 16);
}
__device__ __forceinline__ float sigm(float x) { return 1.0f / (1.0f + __expf(-x)); }
__device__ __forceinline__ float tanh_fast(float x) { return 2.0f * sigm(2.0f * x) - 1.0f; }

// --- (1) pack W[n][k] (f32, [1024,256] row-major) -> Wp[(k/2)*1024 + n] bf16x2;
//     also bsum[dir][n] = bih[n] + bhh[n]
__global__ void wconv_kernel(const float* __restrict__ Wih_f, const float* __restrict__ Whh_f,
                             const float* __restrict__ Wih_b, const float* __restrict__ Whh_b,
                             const float* __restrict__ bih_f, const float* __restrict__ bhh_f,
                             const float* __restrict__ bih_b, const float* __restrict__ bhh_b,
                             unsigned int* __restrict__ Wp, float* __restrict__ bsum) {
  int idx = blockIdx.x * blockDim.x + threadIdx.x;
  if (idx < 4 * PER) {
    int mat = idx / PER, r = idx % PER;
    int kp = r >> 10, n = r & 1023;
    const float* W = (mat == 0) ? Wih_f : (mat == 1) ? Whh_f : (mat == 2) ? Wih_b : Whh_b;
    float lo = W[n * 256 + 2 * kp], hi = W[n * 256 + 2 * kp + 1];
    Wp[idx] = (unsigned int)f2bf(lo) | ((unsigned int)f2bf(hi) << 16);
  } else if (idx < 4 * PER + 2048) {
    int r = idx - 4 * PER, d = r >> 10, n = r & 1023;
    bsum[r] = d == 0 ? (bih_f[n] + bhh_f[n]) : (bih_b[n] + bhh_b[n]);
  }
}

// --- (2) logits[b][t][kk] = b_out[kk]
__global__ void linit_kernel(const float* __restrict__ b_out, float* __restrict__ logits, int n) {
  int i = blockIdx.x * blockDim.x + threadIdx.x;
  if (i < n) logits[i] = b_out[i % 14];
}

// --- (3) recurrent kernel: block = (dir, batch-half), 512 threads = 16 waves
__global__ __launch_bounds__(512) void bilstm_kernel(
    const int*   __restrict__ sent, const float* __restrict__ emb,
    const unsigned int* __restrict__ Wp, const float* __restrict__ bsum,
    const float* __restrict__ h0, const float* __restrict__ c0,
    const float* __restrict__ W_out, float* __restrict__ logits) {
  const int dir  = blockIdx.x >> 1;
  const int bh   = blockIdx.x & 1;
  const int tid  = threadIdx.x;
  const int lane = tid & 31, wv = tid >> 5;

  __shared__ __align__(16) unsigned short sh_x[16 * 264];   // x_t bf16, padded rows
  __shared__ __align__(16) unsigned short sh_h[16 * 264];   // h bf16
  __shared__ __align__(16) float          sh_hf[16 * 260];  // h f32 (for projection)
  __shared__ __align__(16) float          sh_g[16 * 1028];  // gate pre-activations

  const unsigned int* Wih = Wp + dir * 2 * PER;
  const unsigned int* Whh = Wih + PER;

  // warm L2 with this direction's packed weights (global_prefetch_b8)
  for (int off = tid * 64; off < 2 * PER * 4; off += 512 * 64)
    __builtin_prefetch((const char*)Wih + off, 0, 1);

  // init state: c in registers (8 contiguous per thread), h in LDS
  float c[8];
  {
    int base = tid * 8, m = base >> 8, n0e = base & 255;
    const float* c0p = c0 + (dir * 32 + bh * 16 + m) * 256 + n0e;
#pragma unroll
    for (int j = 0; j < 8; ++j) c[j] = c0p[j];
  }
  for (int idx = tid; idx < 4096; idx += 512) {
    int m = idx >> 8, n = idx & 255;
    float hv = h0[(dir * 32 + bh * 16 + m) * 256 + n];
    sh_hf[m * 260 + n] = hv;
    sh_h[m * 264 + n]  = f2bf(hv);
  }

  // per-lane WMMA fragment addressing (constants across the time loop)
  const int m_a   = lane & 15;
  const int arow  = m_a * 132;               // word offset of A row (264 halfwords)
  const int akoff = (lane < 16) ? 0 : 4;     // A: K base 0 / 8 halfwords (in words)
  const int bko   = (lane < 16) ? 0 : 4;     // B: K base 0 / 8 halfwords (in dwords)
  const int mrow  = (lane < 16) ? 0 : 8;     // D: M base row
  const int ncol0 = wv * 64 + m_a;           // first output column for this lane

  const int bias0 = dir * 1024;
  for (int s = 0; s < 1024; ++s) {
    const int t = dir ? (1023 - s) : s;

    // gather + convert embedding rows for this timestep
    for (int idx = tid; idx < 4096; idx += 512) {
      int m = idx >> 8, n = idx & 255;
      int tok = sent[(bh * 16 + m) * 1024 + t];
      sh_x[m * 264 + n] = f2bf(emb[tok * 256 + n]);
    }
    __syncthreads();

    // WMMA GEMM: gates[16 x 1024] = x_t @ Wih^T + h @ Whh^T
    // kb outer (unroll-disabled), A loaded once per k-block, 4 N-tiles inner.
    {
      v8f acc[4];
#pragma unroll
      for (int nt = 0; nt < 4; ++nt) acc[nt] = (v8f){0.f,0.f,0.f,0.f,0.f,0.f,0.f,0.f};

#pragma unroll 1
      for (int pass = 0; pass < 2; ++pass) {
        const unsigned int* aw = pass ? (const unsigned int*)sh_h
                                      : (const unsigned int*)sh_x;
        const unsigned int* wp = pass ? Whh : Wih;
#pragma unroll 1
        for (int kb = 0; kb < 256; kb += 32) {
          FragAB a;
          const int ab = arow + (kb >> 1) + akoff;
#pragma unroll
          for (int v = 0; v < 4; ++v) { a.u[v] = aw[ab + v]; a.u[4 + v] = aw[ab + 8 + v]; }
          const unsigned int* wrow = wp + (size_t)((kb >> 1) + bko) * 1024 + ncol0;
#pragma unroll
          for (int nt = 0; nt < 4; ++nt) {
            FragAB b;
#pragma unroll
            for (int v = 0; v < 8; ++v) b.u[v] = wrow[(v + (v >> 2) * 4) * 1024 + nt * 16];
            acc[nt] = __builtin_amdgcn_wmma_f32_16x16x32_bf16(
                false, a.v, false, b.v, (short)0, acc[nt], false, false);
          }
        }
      }
#pragma unroll
      for (int nt = 0; nt < 4; ++nt)
#pragma unroll
        for (int v = 0; v < 8; ++v)
          sh_g[(mrow + v) * 1028 + ncol0 + nt * 16] = acc[nt][v];
    }
    __syncthreads();

    // fused LSTM cell (gate order i, f, g, o) -- branch-free activations
    {
      int base = tid * 8, m = base >> 8, n0e = base & 255;
      const float* gr = sh_g + m * 1028;
#pragma unroll
      for (int j = 0; j < 8; ++j) {
        int n = n0e + j;
        float ig = gr[n]        + bsum[bias0 + n];
        float fg = gr[256 + n]  + bsum[bias0 + 256 + n];
        float gg = gr[512 + n]  + bsum[bias0 + 512 + n];
        float og = gr[768 + n]  + bsum[bias0 + 768 + n];
        float cn = sigm(fg) * c[j] + sigm(ig) * tanh_fast(gg);
        c[j] = cn;
        float hv = sigm(og) * tanh_fast(cn);
        sh_hf[m * 260 + n] = hv;
        sh_h[m * 264 + n]  = f2bf(hv);
      }
    }
    __syncthreads();

    // fused output projection: logits[b][t][kk] += h . W_out[kk, dir*256:+256]
    if (tid < 224) {
      int b = tid / 14, kk = tid - b * 14;
      const float* wo = W_out + kk * 512 + dir * 256;
      const float* hr = sh_hf + b * 260;
      float sum = 0.f;
#pragma unroll 8
      for (int n = 0; n < 256; ++n) sum += hr[n] * wo[n];
      atomicAdd(&logits[((bh * 16 + b) * 1024 + t) * 14 + kk], sum);
    }
    // no extra barrier needed: next writers of sh_hf are behind two barriers
  }
}

// --- (4) Viterbi: one wave per sequence, lane = tag, backpointers in LDS
__global__ __launch_bounds__(32) void viterbi_kernel(const float* __restrict__ logits,
                                                     const float* __restrict__ crf,
                                                     float* __restrict__ out) {
  const int b = blockIdx.x, lane = threadIdx.x;
  __shared__ float sp[16];
  __shared__ unsigned char sbp[1024 * 16];
  float crfcol[14];
  if (lane < 14)
    for (int f = 0; f < 14; ++f) crfcol[f] = crf[f * 14 + lane];
  const float* lg = logits + b * 1024 * 14;
  float prev = (lane < 14) ? lg[lane] : -3.0e38f;
  for (int t = 1; t < 1024; ++t) {
    if (lane < 16) sp[lane] = prev;
    __syncthreads();
    float best = -3.0e38f; int barg = 0;
    if (lane < 14) {
#pragma unroll
      for (int f = 0; f < 14; ++f) {
        float v = sp[f] + crfcol[f];
        if (v > best) { best = v; barg = f; }   // strict > keeps first max (argmax semantics)
      }
      prev = lg[t * 14 + lane] + best;
    }
    sbp[t * 16 + lane] = (unsigned char)barg;
    __syncthreads();
  }
  if (lane < 16) sp[lane] = prev;
  __syncthreads();
  if (lane == 0) {
    float best = sp[0]; int tag = 0;
    for (int k = 1; k < 14; ++k) if (sp[k] > best) { best = sp[k]; tag = k; }
    out[b] = best;
    float* path = out + 32 + b * 1024;
    path[1023] = (float)tag;
    for (int t = 1023; t >= 1; --t) {
      tag = sbp[t * 16 + tag];
      path[t - 1] = (float)tag;
    }
  }
}

extern "C" void kernel_launch(void* const* d_in, const int* in_sizes, int n_in,
                              void* d_out, int out_size, void* d_ws, size_t ws_size,
                              hipStream_t stream) {
  const int*   sent  = (const int*)d_in[0];
  const float* emb   = (const float*)d_in[1];
  const float* Wih_f = (const float*)d_in[2];
  const float* Whh_f = (const float*)d_in[3];
  const float* bih_f = (const float*)d_in[4];
  const float* bhh_f = (const float*)d_in[5];
  const float* Wih_b = (const float*)d_in[6];
  const float* Whh_b = (const float*)d_in[7];
  const float* bih_b = (const float*)d_in[8];
  const float* bhh_b = (const float*)d_in[9];
  const float* h0    = (const float*)d_in[10];
  const float* c0    = (const float*)d_in[11];
  const float* W_out = (const float*)d_in[12];
  const float* b_out = (const float*)d_in[13];
  const float* crf   = (const float*)d_in[14];
  float* out = (float*)d_out;

  char* ws = (char*)d_ws;
  unsigned int* Wp     = (unsigned int*)ws;                            // 2 MiB packed weights
  float*        bsum   = (float*)(ws + (size_t)4 * PER * 4);           // 8 KiB bias sums
  float*        logits = (float*)(ws + (size_t)4 * PER * 4 + 2048 * 4);// 1.75 MiB

  wconv_kernel<<<(4 * PER + 2048) / 256, 256, 0, stream>>>(
      Wih_f, Whh_f, Wih_b, Whh_b, bih_f, bhh_f, bih_b, bhh_b, Wp, bsum);
  linit_kernel<<<NLOG / 256, 256, 0, stream>>>(b_out, logits, NLOG);
  bilstm_kernel<<<4, 512, 0, stream>>>(sent, emb, Wp, bsum, h0, c0, W_out, logits);
  viterbi_kernel<<<32, 32, 0, stream>>>(logits, crf, out);
}
